// KnowledgeGraphGNN_41489384080024
// MI455X (gfx1250) — matrix operand order repaired and checked by reading, not compiled, not run on hardware
//
#include <hip/hip_runtime.h>
#include <math.h>

typedef __attribute__((ext_vector_type(16))) _Float16 v16h;
typedef __attribute__((ext_vector_type(8)))  float    v8f;

// ---------------------------------------------------------------- degree prep
__global__ void k_deg_init(float* __restrict__ deg, int n) {
    int i = blockIdx.x * blockDim.x + threadIdx.x;
    if (i < n) deg[i] = 1.0f;                      // self-loop contributes 1
}

__global__ void k_deg_accum(const int* __restrict__ dst, float* __restrict__ deg, int nE) {
    int e = blockIdx.x * blockDim.x + threadIdx.x;
    if (e < nE) atomicAdd(&deg[dst[e]], 1.0f);
}

__global__ void k_dinv(float* __restrict__ deg, int n) {
    int i = blockIdx.x * blockDim.x + threadIdx.x;
    if (i < n) {
        float d = deg[i];
        deg[i] = (d > 0.0f) ? rsqrtf(d) : 0.0f;    // deg >= 1 always (self loop)
    }
}

// ---------------------------------------------------------------- weight pack
// Repack W[K,N] f32 row-major into f16 WMMA B-fragment order:
//   Wp[((nt*(K/32) + ks)*32 + lane)*16 + e]  where
//   n = nt*16 + (lane&15), k = ks*32 + (lane>>4)*16 + e
// so the GEMM reads each 16x32 B fragment as one contiguous 32B v16h per lane.
__global__ void k_pack_w(const float* __restrict__ W, _Float16* __restrict__ Wp,
                         int K, int N)
{
    int i = blockIdx.x * blockDim.x + threadIdx.x;
    if (i >= K * N) return;
    int k = i / N, n1 = i % N;
    int nt   = n1 >> 4;
    int ks   = k >> 5;
    int lane = (n1 & 15) | (((k >> 4) & 1) << 4);
    int e    = k & 15;
    Wp[((size_t)((nt * (K >> 5) + ks) * 32 + lane) << 4) + e] = (_Float16)W[i];
}

// ---------------------------------------------------------------- WMMA GEMM
// C[M,N] = A[M,K] @ B[K,N].  One wave per 16x64 output slab: 4 accumulators,
// 4 independent v_wmma per K-step (covers the f16 WMMA RAW hazard distance),
// K fully unrolled via template.  B comes pre-packed in fragment order.
template <int K>
__global__ __launch_bounds__(256) void k_gemm_wmma_p(
    const float* __restrict__ A, const _Float16* __restrict__ Bp,
    float* __restrict__ C, int M, int N)
{
    const int lane = threadIdx.x & 31;
    const int wv   = threadIdx.x >> 5;
    const int mt   = blockIdx.y * 8 + wv;       // 16-row tile index
    const int m0   = mt * 16;
    if (m0 >= M) return;
    const int n0   = blockIdx.x * 64;           // 64-column slab

    const int t  = lane & 15;
    const int hf = lane >> 4;

    int mrow = m0 + t;
    if (mrow >= M) mrow = M - 1;                // clamp (fast path below unguarded)
    const float* __restrict__ arow = A + (size_t)mrow * K;

    constexpr int KS = K / 32;
    const int nt0 = n0 >> 4;
    const _Float16* __restrict__ bbase = Bp + ((size_t)lane << 4);

    v8f a0 = {}, a1 = {}, a2 = {}, a3 = {};
#pragma unroll
    for (int ks = 0; ks < KS; ++ks) {
        const int k0 = ks * 32;
        v16h a;
#pragma unroll
        for (int e = 0; e < 8; ++e) {
            a[e]     = (_Float16)arow[k0 + hf * 16 + e];
            a[e + 8] = (_Float16)arow[k0 + hf * 16 + 8 + e];
        }
        const v16h b0 = *(const v16h*)(bbase + ((size_t)((nt0 + 0) * KS + ks) << 9));
        const v16h b1 = *(const v16h*)(bbase + ((size_t)((nt0 + 1) * KS + ks) << 9));
        const v16h b2 = *(const v16h*)(bbase + ((size_t)((nt0 + 2) * KS + ks) << 9));
        const v16h b3 = *(const v16h*)(bbase + ((size_t)((nt0 + 3) * KS + ks) << 9));
        a0 = __builtin_amdgcn_wmma_f32_16x16x32_f16(false, a, false, b0, (short)0, a0, false, false);
        a1 = __builtin_amdgcn_wmma_f32_16x16x32_f16(false, a, false, b1, (short)0, a1, false, false);
        a2 = __builtin_amdgcn_wmma_f32_16x16x32_f16(false, a, false, b2, (short)0, a2, false, false);
        a3 = __builtin_amdgcn_wmma_f32_16x16x32_f16(false, a, false, b3, (short)0, a3, false, false);
    }

    // C/D layout: VGPR r -> row m0 + r + 8*hf, col n0 + 16*tile + t
    if (m0 + 16 <= M) {                         // wave-uniform fast path
#pragma unroll
        for (int r = 0; r < 8; ++r) {
            float* __restrict__ crow = C + (size_t)(m0 + r + 8 * hf) * N + n0 + t;
            crow[0]  = a0[r];
            crow[16] = a1[r];
            crow[32] = a2[r];
            crow[48] = a3[r];
        }
    } else {
#pragma unroll
        for (int r = 0; r < 8; ++r) {
            int mr = m0 + r + 8 * hf;
            if (mr < M) {
                float* __restrict__ crow = C + (size_t)mr * N + n0 + t;
                crow[0]  = a0[r];
                crow[16] = a1[r];
                crow[32] = a2[r];
                crow[48] = a3[r];
            }
        }
    }
}

// Generic fallback (runtime K, one 16x16 tile per wave, raw f32 weights).
__global__ __launch_bounds__(256) void k_gemm_generic(
    const float* __restrict__ A, const float* __restrict__ B,
    float* __restrict__ C, int M, int K, int N)
{
    const int lane = threadIdx.x & 31;
    const int wv   = threadIdx.x >> 5;
    const int n0   = (blockIdx.x * 8 + wv) * 16;
    const int m0   = blockIdx.y * 16;
    if (n0 >= N) return;
    const int t  = lane & 15;
    const int hf = lane >> 4;
    int mrow = m0 + t;
    if (mrow >= M) mrow = M - 1;
    const float* __restrict__ arow = A + (size_t)mrow * K;

    v8f acc = {};
    for (int k0 = 0; k0 + 32 <= K; k0 += 32) {
        v16h a, b;
#pragma unroll
        for (int e = 0; e < 8; ++e) {
            a[e]     = (_Float16)arow[k0 + hf * 16 + e];
            a[e + 8] = (_Float16)arow[k0 + hf * 16 + 8 + e];
        }
        const float* __restrict__ bp = B + (size_t)(k0 + hf * 16) * N + n0 + t;
#pragma unroll
        for (int e = 0; e < 16; ++e)
            b[e] = (_Float16)bp[(size_t)e * N];
        acc = __builtin_amdgcn_wmma_f32_16x16x32_f16(false, a, false, b, (short)0, acc, false, false);
    }
    if (m0 + 16 <= M) {
#pragma unroll
        for (int r = 0; r < 8; ++r)
            C[(size_t)(m0 + r + 8 * hf) * N + n0 + t] = acc[r];
    } else {
#pragma unroll
        for (int r = 0; r < 8; ++r) {
            int mr = m0 + r + 8 * hf;
            if (mr < M) C[(size_t)mr * N + n0 + t] = acc[r];
        }
    }
}

// -------------------------------------------------- node init: bias + self loop
// out[node,c] = bias[c] + dinv[node]^2 * xw[node,c]   (2D grid: y = node)
__global__ void k_node_init(const float* __restrict__ xw, const float* __restrict__ dinv,
                            const float* __restrict__ bias, float* __restrict__ out,
                            int C)
{
    int c = blockIdx.x * blockDim.x + threadIdx.x;
    if (c >= C) return;
    int node = blockIdx.y;
    float dv = dinv[node];
    size_t i = (size_t)node * C + c;
    out[i] = bias[c] + dv * dv * xw[i];
}

// -------------------------------------------------- edge gather/scale/scatter
__global__ __launch_bounds__(256) void k_edge_scatter(
    const float* __restrict__ xw, const int* __restrict__ src,
    const int* __restrict__ dst, const float* __restrict__ dinv,
    float* __restrict__ out, int C, int nE)
{
    int e = blockIdx.x * 8 + (threadIdx.x >> 5);
    if (e >= nE) return;
    int lane = threadIdx.x & 31;
    int s = src[e], d = dst[e];
    float nrm = dinv[s] * dinv[d];
    const float* __restrict__ xs = xw + (size_t)s * C;
    float* __restrict__ od = out + (size_t)d * C;
    for (int c = lane * 4; c < C; c += 128) {
        const float4 v = *(const float4*)(xs + c);
        atomicAdd(od + c + 0, nrm * v.x);
        atomicAdd(od + c + 1, nrm * v.y);
        atomicAdd(od + c + 2, nrm * v.z);
        atomicAdd(od + c + 3, nrm * v.w);
    }
}

__global__ void k_relu(float* __restrict__ x, size_t n) {
    size_t i = (size_t)blockIdx.x * blockDim.x + threadIdx.x;
    if (i < n) x[i] = fmaxf(x[i], 0.0f);
}

// ---------------------------------------------------------------- pooling
__global__ void k_zero(float* __restrict__ p, int n) {
    int i = blockIdx.x * blockDim.x + threadIdx.x;
    if (i < n) p[i] = 0.0f;
}

__global__ void k_pool(const float* __restrict__ h, const int* __restrict__ batch,
                       float* __restrict__ sums, float* __restrict__ cnt, int C)
{
    int c = blockIdx.x * blockDim.x + threadIdx.x;
    if (c >= C) return;
    int node = blockIdx.y;
    int b    = batch[node];
    atomicAdd(&sums[(size_t)b * C + c], h[(size_t)node * C + c]);
    if (c == 0) atomicAdd(&cnt[b], 1.0f);
}

__global__ void k_pool_div(float* __restrict__ sums, const float* __restrict__ cnt,
                           int G, int C)
{
    int i = blockIdx.x * blockDim.x + threadIdx.x;
    if (i < G * C) sums[i] /= fmaxf(cnt[i / C], 1.0f);
}

// ---------------------------------------------------------------- GEMM dispatch
static void launch_gemm(const float* A, const _Float16* Bp, const float* Braw,
                        float* C, int M, int K, int N, hipStream_t stream)
{
    if ((N % 64) == 0 && K == 128) {
        dim3 g((unsigned)(N / 64), (unsigned)((M / 16 + 8) / 8));
        k_gemm_wmma_p<128><<<g, 256, 0, stream>>>(A, Bp, C, M, N);
    } else if ((N % 64) == 0 && K == 256) {
        dim3 g((unsigned)(N / 64), (unsigned)((M / 16 + 8) / 8));
        k_gemm_wmma_p<256><<<g, 256, 0, stream>>>(A, Bp, C, M, N);
    } else {
        dim3 g((unsigned)((N + 127) / 128), (unsigned)((M + 15) / 16));
        k_gemm_generic<<<g, 256, 0, stream>>>(A, Braw, C, M, K, N);
    }
}

// ---------------------------------------------------------------- launch
extern "C" void kernel_launch(void* const* d_in, const int* in_sizes, int n_in,
                              void* d_out, int out_size, void* d_ws, size_t ws_size,
                              hipStream_t stream)
{
    (void)n_in; (void)ws_size;
    const float* X     = (const float*)d_in[0];  // [n, IN]
    const int*   EI    = (const int*)  d_in[1];  // [2, E]
    const int*   batch = (const int*)  d_in[2];  // [n]
    const float* W1    = (const float*)d_in[3];  // [IN, HID]
    const float* b1    = (const float*)d_in[4];  // [HID]
    const float* W2    = (const float*)d_in[5];  // [HID, OC]
    const float* b2    = (const float*)d_in[6];  // [OC]
    float* OUT = (float*)d_out;                  // [G, OC]

    const int n   = in_sizes[2];
    const int E   = in_sizes[1] / 2;
    const int IN  = in_sizes[0] / n;
    const int HID = in_sizes[4];
    const int OC  = in_sizes[6];
    const int G   = out_size / OC;

    const int* src = EI;
    const int* dst = EI + E;

    // workspace carve-up (256B aligned)
    char* ws = (char*)d_ws;
    size_t off = 0;
    auto aln = [](size_t x) { return (x + 255) & ~(size_t)255; };
    float*     dinv = (float*)(ws + off);     off = aln(off + (size_t)n * 4);
    float*     xwB  = (float*)(ws + off);     off = aln(off + (size_t)n * HID * 4);
    float*     hB   = (float*)(ws + off);     off = aln(off + (size_t)n * HID * 4);
    _Float16*  W1p  = (_Float16*)(ws + off);  off = aln(off + (size_t)IN * HID * 2);
    _Float16*  W2p  = (_Float16*)(ws + off);  off = aln(off + (size_t)HID * OC * 2);
    float*     cnt  = (float*)(ws + off);     off = aln(off + (size_t)G * 4);

    // ---- degrees / symmetric norm + weight packing (independent prep)
    k_deg_init <<<(n + 255) / 256, 256, 0, stream>>>(dinv, n);
    k_deg_accum<<<(E + 255) / 256, 256, 0, stream>>>(dst, dinv, E);
    k_dinv     <<<(n + 255) / 256, 256, 0, stream>>>(dinv, n);
    k_pack_w   <<<(IN * HID + 255) / 256, 256, 0, stream>>>(W1, W1p, IN, HID);
    k_pack_w   <<<(HID * OC + 255) / 256, 256, 0, stream>>>(W2, W2p, HID, OC);

    // ---- layer 1: XW1 = X @ W1 ; H1 = relu(scatter(norm * XW1) + b1)
    launch_gemm(X, W1p, W1, xwB, n, IN, HID, stream);
    size_t nc1 = (size_t)n * HID;
    {
        dim3 g((unsigned)((HID + 255) / 256), (unsigned)n);
        k_node_init<<<g, 256, 0, stream>>>(xwB, dinv, b1, hB, HID);
    }
    k_edge_scatter<<<(E + 7) / 8, 256, 0, stream>>>(xwB, src, dst, dinv, hB, HID, E);
    k_relu<<<(unsigned)((nc1 + 255) / 256), 256, 0, stream>>>(hB, nc1);

    // ---- layer 2: XW2 = H1 @ W2 ; H2 = relu(scatter(norm * XW2) + b2)
    launch_gemm(hB, W2p, W2, xwB, n, HID, OC, stream);
    size_t nc2 = (size_t)n * OC;
    {
        dim3 g((unsigned)((OC + 255) / 256), (unsigned)n);
        k_node_init<<<g, 256, 0, stream>>>(xwB, dinv, b2, hB, OC);
    }
    k_edge_scatter<<<(E + 7) / 8, 256, 0, stream>>>(xwB, src, dst, dinv, hB, OC, E);
    k_relu<<<(unsigned)((nc2 + 255) / 256), 256, 0, stream>>>(hB, nc2);

    // ---- global mean pool over batch
    k_zero<<<(G * OC + 255) / 256, 256, 0, stream>>>(OUT, G * OC);
    k_zero<<<(G + 255) / 256, 256, 0, stream>>>(cnt, G);
    {
        dim3 g((unsigned)((OC + 255) / 256), (unsigned)n);
        k_pool<<<g, 256, 0, stream>>>(hB, batch, OUT, cnt, OC);
    }
    k_pool_div<<<(G * OC + 255) / 256, 256, 0, stream>>>(OUT, cnt, G, OC);
}
